// KDSR_15616501088880
// MI455X (gfx1250) — matrix operand order repaired and checked by reading, not compiled, not run on hardware
//
#include <hip/hip_runtime.h>
#include <hip/hip_bf16.h>

// ---------------- problem constants ----------------
#define Bc 1024
#define Nc 24
#define Lc 24
#define Dc 128
#define Vc 40000
#define NEGC (-9e15f)

// ---------------- WMMA vector types ----------------
typedef __attribute__((ext_vector_type(16))) __bf16 v16bf;
typedef __attribute__((ext_vector_type(8)))  float  v8f;
typedef __attribute__((ext_vector_type(4)))  float  f32x4;

union BFV { v16bf v; f32x4 q[2]; };

// round-to-nearest-even fp32 -> bf16 (bit trick; NaN edge ignored)
__device__ __forceinline__ unsigned short f2bf(float f) {
  unsigned u = __float_as_uint(f);
  unsigned r = u + 0x7FFFu + ((u >> 16) & 1u);
  return (unsigned short)(r >> 16);
}

// A operand, 16-bit 16x32 layout: lane l<16 -> row l, elems[0..7]=K(kb+0..7),
// elems[8..15]=K(kb+16..23); lane l>=16 -> same row, K offset +8.
__device__ __forceinline__ v16bf loadA(const unsigned short* rowbase, int kbase, int half) {
  BFV u;
  u.q[0] = *(const f32x4*)(rowbase + kbase + half * 8);
  u.q[1] = *(const f32x4*)(rowbase + kbase + 16 + half * 8);
  return u.v;
}

// B operand, 16-bit 32x16 layout: lane l<16 -> col l, elems[0..7]=K(kb+0..7),
// elems[8..15]=K(kb+16..23); lane l>=16 -> col l-16, K offset +8. colbase must be K-contiguous.
__device__ __forceinline__ v16bf loadB(const unsigned short* colbase, int kbase, int half) {
  BFV u;
  u.q[0] = *(const f32x4*)(colbase + kbase + half * 8);
  u.q[1] = *(const f32x4*)(colbase + kbase + 16 + half * 8);
  return u.v;
}

#define WMMA_BF16(a, b, c) \
  __builtin_amdgcn_wmma_f32_16x16x32_bf16(false, (a), false, (b), (short)0, (c), false, false)

// ---------------- kernel 1: fp32 tables -> bf16 ----------------
__global__ __launch_bounds__(256) void k_convert(const float* __restrict__ a,
                                                 const float* __restrict__ b,
                                                 const float* __restrict__ c,
                                                 unsigned short* __restrict__ ab,
                                                 unsigned short* __restrict__ bb,
                                                 unsigned short* __restrict__ cb) {
  int i = blockIdx.x * 256 + threadIdx.x;
  if (i < Vc * Dc) {
    ab[i] = f2bf(a[i]);
    bb[i] = f2bf(b[i]);
    cb[i] = f2bf(c[i]);
  }
}

// ---------------- kernel 2: fold fusion weights ----------------
// Mt_item[n][k] = Wf0[k][n]
// Mt_img [n][k] = (W_img @ Wf1)[k][n],  Mt_txt[n][k] = (W_txt @ Wf2)[k][n]
// Stored transposed (n-major) so WMMA B loads are K-contiguous.
__global__ __launch_bounds__(128) void k_combine(const float* __restrict__ W_img,
                                                 const float* __restrict__ W_txt,
                                                 const float* __restrict__ W_fuse,
                                                 unsigned short* __restrict__ wt) {
  int mat = blockIdx.y, k = blockIdx.x, n = threadIdx.x;
  float v;
  if (mat == 0) {
    v = W_fuse[k * Dc + n];
  } else {
    const float* W = (mat == 1) ? W_img : W_txt;
    int ro = (mat == 1) ? Dc : 2 * Dc;
    float s = 0.f;
    for (int t = 0; t < Dc; ++t) s += W[k * Dc + t] * W_fuse[(ro + t) * Dc + n];
    v = s;
  }
  wt[mat * Dc * Dc + n * Dc + k] = f2bf(v);
}

__global__ __launch_bounds__(128) void k_cbias(const float* __restrict__ W_fuse,
                                               const float* __restrict__ b_img,
                                               const float* __restrict__ b_txt,
                                               const float* __restrict__ b_fuse,
                                               float* __restrict__ cb) {
  int n = threadIdx.x;
  float s = b_fuse[n];
  for (int t = 0; t < Dc; ++t)
    s += b_img[t] * W_fuse[(Dc + t) * Dc + n] + b_txt[t] * W_fuse[(2 * Dc + t) * Dc + n];
  cb[n] = s;
}

// ---------------- kernel 3: fused multi-modal projection (WMMA) ----------------
// h[row] = e_item@Wf0 + e_img@M_img + e_txt@M_txt + c  for row in [0, B*N)
// One wave per 16-row tile; 8 column tiles x 3 matrices x 4 ksteps = 96 WMMA.
__global__ __launch_bounds__(256) void k_fuse(const int* __restrict__ items,
                                              const unsigned short* __restrict__ embbf,
                                              const unsigned short* __restrict__ imgbf,
                                              const unsigned short* __restrict__ txtbf,
                                              const unsigned short* __restrict__ wts,
                                              const float* __restrict__ cbias,
                                              float* __restrict__ h) {
  const int lane = threadIdx.x & 31, wid = threadIdx.x >> 5;
  const int tile = blockIdx.x * 8 + wid;  // 1536 tiles exactly
  const int rl = lane & 15, half = lane >> 4;
  const int row = tile * 16 + rl;
  const int item = items[row];

  const unsigned short* tabs[3] = {embbf + (size_t)item * Dc,
                                   imgbf + (size_t)item * Dc,
                                   txtbf + (size_t)item * Dc};

  for (int nt = 0; nt < 8; ++nt) {
    const int nb = nt * 16;
    v8f c = {};
#pragma unroll
    for (int m = 0; m < 3; ++m) {
      const unsigned short* ar = tabs[m];
      const unsigned short* wc = wts + m * Dc * Dc + (size_t)(nb + rl) * Dc;
#pragma unroll
      for (int k4 = 0; k4 < 4; ++k4) {
        v16bf a = loadA(ar, k4 * 32, half);
        v16bf b = loadB(wc, k4 * 32, half);
        c = WMMA_BF16(a, b, c);
      }
    }
    const float cb = cbias[nb + rl];
#pragma unroll
    for (int v = 0; v < 8; ++v)
      h[(size_t)(tile * 16 + v + half * 8) * Dc + nb + rl] = c[v] + cb;
  }
}

// ---------------- kernel 4: RGAT + readout, one block per batch ----------------
__global__ __launch_bounds__(256) void k_rgat(const float* __restrict__ h,
                                              const int* __restrict__ adj,
                                              const int* __restrict__ alias_,
                                              const int* __restrict__ mask_,
                                              const float* __restrict__ a_rel,
                                              const float* __restrict__ pos_emb,
                                              const float* __restrict__ w_1,
                                              const float* __restrict__ w_2,
                                              const float* __restrict__ glu1_w,
                                              const float* __restrict__ glu1_b,
                                              const float* __restrict__ glu2_w,
                                              unsigned short* __restrict__ selbf) {
  const int b = blockIdx.x, tid = threadIdx.x;
  __shared__ float sh[Nc * Dc];    // node features
  __shared__ float sar[4 * Dc];    // relation vectors
  __shared__ float satt[Nc * Nc];  // logits -> attention
  __shared__ float sagg[Nc * Dc];  // aggregated nodes (reused as gate)
  __shared__ float sseq[Nc * Dc];  // sequence-ordered
  __shared__ float snh[Nc * Dc];   // tanh features
  __shared__ float shs[Dc], shsg[Dc], sbeta[Nc], smask[Nc];

  const float* hb = h + (size_t)b * Nc * Dc;
  for (int i = tid; i < Nc * Dc; i += 256) sh[i] = hb[i];
  for (int i = tid; i < 4 * Dc; i += 256) sar[i] = a_rel[i];
  if (tid < Nc) smask[tid] = (float)mask_[b * Lc + tid];
  __syncthreads();

  // relation-typed scores + leaky-relu + adjacency masking
  for (int p = tid; p < Nc * Nc; p += 256) {
    int i = p / Nc, j = p % Nc;
    const float* hi = &sh[i * Dc];
    const float* hj = &sh[j * Dc];
    float s0 = 0, s1 = 0, s2 = 0, s3 = 0;
    for (int d = 0; d < Dc; ++d) {
      float pr = hi[d] * hj[d];
      s0 += pr * sar[d];
      s1 += pr * sar[Dc + d];
      s2 += pr * sar[2 * Dc + d];
      s3 += pr * sar[3 * Dc + d];
    }
    int a = adj[(size_t)b * Nc * Nc + p];
    float e = (a == 1) ? s0 : (a == 2) ? s1 : (a == 3) ? s2 : s3;
    e = (e > 0.f) ? e : 0.2f * e;
    satt[p] = (a >= 1 && a <= 4) ? e : NEGC;
  }
  __syncthreads();

  // row softmax
  if (tid < Nc) {
    float mx = -3.4e38f;
    for (int j = 0; j < Nc; ++j) mx = fmaxf(mx, satt[tid * Nc + j]);
    float s = 0.f;
    for (int j = 0; j < Nc; ++j) {
      float v = __expf(satt[tid * Nc + j] - mx);
      satt[tid * Nc + j] = v;
      s += v;
    }
    float inv = 1.f / s;
    for (int j = 0; j < Nc; ++j) satt[tid * Nc + j] *= inv;
  }
  __syncthreads();

  // aggregate
  for (int q = tid; q < Nc * Dc; q += 256) {
    int i = q >> 7, d = q & 127;
    float s = 0.f;
    for (int j = 0; j < Nc; ++j) s += satt[i * Nc + j] * sh[j * Dc + d];
    sagg[q] = s;
  }
  __syncthreads();

  // alias gather to sequence order
  for (int q = tid; q < Lc * Dc; q += 256) {
    int l = q >> 7, d = q & 127;
    sseq[q] = sagg[alias_[b * Lc + l] * Dc + d];
  }
  __syncthreads();

  // masked mean
  if (tid < Dc) {
    float s = 0.f, cnt = 0.f;
    for (int l = 0; l < Lc; ++l) {
      s += smask[l] * sseq[l * Dc + tid];
      cnt += smask[l];
    }
    shs[tid] = s / fmaxf(cnt, 1.f);
  }
  __syncthreads();

  // nh = tanh([pos_rev, seq] @ w_1)
  for (int q = tid; q < Lc * Dc; q += 256) {
    int l = q >> 7, d = q & 127;
    const float* pr = pos_emb + (size_t)(Lc - 1 - l) * Dc;
    float s = 0.f;
    for (int t = 0; t < Dc; ++t) s += pr[t] * w_1[t * Dc + d];
    for (int t = 0; t < Dc; ++t) s += sseq[l * Dc + t] * w_1[(Dc + t) * Dc + d];
    snh[q] = tanhf(s);
  }
  // hsg = hs @ glu2_w
  if (tid < Dc) {
    float s = 0.f;
    for (int t = 0; t < Dc; ++t) s += shs[t] * glu2_w[t * Dc + tid];
    shsg[tid] = s;
  }
  __syncthreads();

  // gate = sigmoid(nh @ glu1_w + glu1_b + hsg)  (reuse sagg)
  for (int q = tid; q < Lc * Dc; q += 256) {
    int l = q >> 7, d = q & 127;
    float s = 0.f;
    for (int t = 0; t < Dc; ++t) s += snh[l * Dc + t] * glu1_w[t * Dc + d];
    s += glu1_b[d] + shsg[d];
    sagg[q] = 1.f / (1.f + __expf(-s));
  }
  __syncthreads();

  // beta = gate @ w_2
  if (tid < Lc) {
    float s = 0.f;
    for (int d = 0; d < Dc; ++d) s += sagg[tid * Dc + d] * w_2[d];
    sbeta[tid] = s;
  }
  __syncthreads();

  // select = sum_l beta*m*seq  -> bf16 row for the scoring GEMM
  if (tid < Dc) {
    float s = 0.f;
    for (int l = 0; l < Lc; ++l) s += sbeta[l] * smask[l] * sseq[l * Dc + tid];
    selbf[(size_t)b * Dc + tid] = f2bf(s);
  }
}

// ---------------- kernel 5: scores = select @ emb[1:].T (WMMA, 164MB store-bound) ----------------
__global__ __launch_bounds__(256) void k_scores(const unsigned short* __restrict__ selbf,
                                                const unsigned short* __restrict__ embbf,
                                                float* __restrict__ out) {
  const int lane = threadIdx.x & 31, wid = threadIdx.x >> 5;
  const int rl = lane & 15, half = lane >> 4;
  const int mbase = blockIdx.y * 16;
  const int ntb = blockIdx.x * 32 + wid * 4;  // 4 N-tiles per wave

  // cache A tile (16 rows x 128 K) in VGPRs across the N loop
  const unsigned short* ar = selbf + (size_t)(mbase + rl) * Dc;
  v16bf a0 = loadA(ar, 0, half);
  v16bf a1 = loadA(ar, 32, half);
  v16bf a2 = loadA(ar, 64, half);
  v16bf a3 = loadA(ar, 96, half);

  for (int t = 0; t < 4; ++t) {
    const int ntile = ntb + t;          // uniform per wave
    if (ntile >= 2500) break;           // ceil(39999/16) = 2500
    const int cr = ntile * 16 + rl;     // output column
    int er = cr + 1;                    // emb row (pad item 0 excluded)
    if (er > Vc - 1) er = Vc - 1;       // clamp tail loads
    const unsigned short* bc = embbf + (size_t)er * Dc;
    __builtin_prefetch(bc + 16 * Dc, 0, 0);  // global_prefetch next column tile
    v8f c = {};
    c = WMMA_BF16(a0, loadB(bc, 0, half), c);
    c = WMMA_BF16(a1, loadB(bc, 32, half), c);
    c = WMMA_BF16(a2, loadB(bc, 64, half), c);
    c = WMMA_BF16(a3, loadB(bc, 96, half), c);
    if (cr < Vc - 1) {  // guard after WMMA: EXEC all-ones during matrix op
#pragma unroll
      for (int v = 0; v < 8; ++v)
        out[(size_t)(mbase + v + half * 8) * (Vc - 1) + cr] = c[v];
    }
  }
}

// ---------------- launch ----------------
extern "C" void kernel_launch(void* const* d_in, const int* in_sizes, int n_in,
                              void* d_out, int out_size, void* d_ws, size_t ws_size,
                              hipStream_t stream) {
  const int* items = (const int*)d_in[0];
  const int* adj = (const int*)d_in[1];
  const int* alias_ = (const int*)d_in[2];
  const int* mask_ = (const int*)d_in[3];
  const float* emb = (const float*)d_in[4];
  const float* img = (const float*)d_in[5];
  const float* txt = (const float*)d_in[6];
  const float* W_img = (const float*)d_in[7];
  const float* b_img = (const float*)d_in[8];
  const float* W_txt = (const float*)d_in[9];
  const float* b_txt = (const float*)d_in[10];
  const float* W_fuse = (const float*)d_in[11];
  const float* b_fuse = (const float*)d_in[12];
  const float* a_rel = (const float*)d_in[13];
  const float* pos = (const float*)d_in[14];
  const float* w_1 = (const float*)d_in[15];
  const float* w_2 = (const float*)d_in[16];
  const float* glu1_w = (const float*)d_in[17];
  const float* glu1_b = (const float*)d_in[18];
  const float* glu2_w = (const float*)d_in[19];
  float* out = (float*)d_out;

  // workspace layout (all offsets 256B aligned), total ~43.7 MB
  char* ws = (char*)d_ws;
  const size_t tab = (size_t)Vc * Dc * sizeof(unsigned short);  // 10,240,000
  unsigned short* embbf = (unsigned short*)(ws);
  unsigned short* imgbf = (unsigned short*)(ws + tab);
  unsigned short* txtbf = (unsigned short*)(ws + 2 * tab);
  unsigned short* wts = (unsigned short*)(ws + 3 * tab);                    // 3*128*128 bf16
  float* cbias = (float*)(ws + 3 * tab + 3 * Dc * Dc * 2);                  // 128 f32
  float* h = (float*)(ws + 3 * tab + 3 * Dc * Dc * 2 + 512);                // B*N*D f32
  unsigned short* selbf = (unsigned short*)((char*)h + (size_t)Bc * Nc * Dc * 4);

  k_convert<<<(Vc * Dc) / 256, 256, 0, stream>>>(emb, img, txt, embbf, imgbf, txtbf);
  k_combine<<<dim3(Dc, 3), Dc, 0, stream>>>(W_img, W_txt, W_fuse, wts);
  k_cbias<<<1, Dc, 0, stream>>>(W_fuse, b_img, b_txt, b_fuse, cbias);
  k_fuse<<<(Bc * Nc / 16) / 8, 256, 0, stream>>>(items, embbf, imgbf, txtbf, wts, cbias, h);
  k_rgat<<<Bc, 256, 0, stream>>>(h, adj, alias_, mask_, a_rel, pos, w_1, w_2,
                                 glu1_w, glu1_b, glu2_w, selbf);
  k_scores<<<dim3(79, Bc / 16), 256, 0, stream>>>(selbf, embbf, out);
}
